// DGCNN_39711267619402
// MI455X (gfx1250) — matrix operand order repaired and checked by reading, not compiled, not run on hardware
//
#include <hip/hip_runtime.h>

// ---------------------------------------------------------------------------
// DGCNN block for MI455X (gfx1250, wave32, WMMA bf16 16x16x32)
// ---------------------------------------------------------------------------
constexpr int   cB   = 8;
constexpr int   cN   = 4096;
constexpr int   cK   = 20;            // knn neighbours
constexpr int   NKp  = cN * cK;       // 81920 (n,k) positions per batch
constexpr float EPSc = 1e-5f;
constexpr float SLP  = 0.2f;

typedef __bf16  bf16_t;
typedef bf16_t  v16bf __attribute__((ext_vector_type(16)));
typedef float   v8f   __attribute__((ext_vector_type(8)));

struct alignas(16) U4   { unsigned int w[4]; };
struct            U4x2 { U4 a, b; };
struct            H16  { unsigned short h[16]; };

__device__ __forceinline__ unsigned short f2bf(float f) {
  unsigned u = __builtin_bit_cast(unsigned, f);
  u += 0x7FFFu + ((u >> 16) & 1u);            // round-to-nearest-even
  return (unsigned short)(u >> 16);
}
__device__ __forceinline__ float bf2f(unsigned short h) {
  unsigned u = ((unsigned)h) << 16;
  return __builtin_bit_cast(float, u);
}
// Load a 16-element bf16 fragment from two 16B-aligned 8-element chunks.
__device__ __forceinline__ v16bf ld_frag16(const unsigned short* p0,
                                           const unsigned short* p1) {
  U4x2 s;
  s.a = *(const U4*)p0;
  s.b = *(const U4*)p1;
  return __builtin_bit_cast(v16bf, s);
}
__device__ __forceinline__ v8f wmma_bf16(v16bf a, v16bf b, v8f c) {
  return __builtin_amdgcn_wmma_f32_16x16x32_bf16(
      /*neg_a=*/false, a, /*neg_b=*/false, b,
      /*c_mod=*/(short)0, c, /*reuse_a=*/false, /*reuse_b=*/false);
}

// ---------------------------------------------------------------------------
// prep: x (f32 [b][64][n]) -> catb channels 0..63 (bf16, point-major, stride
// 192) + squared norms.  Also converts all weights to bf16.
// ---------------------------------------------------------------------------
__global__ __launch_bounds__(256) void prep_x_kernel(
    const float* __restrict__ x, unsigned short* __restrict__ catb,
    float* __restrict__ sq1) {
  int id = blockIdx.x * 256 + threadIdx.x;           // (b,n)
  if (id >= cB * cN) return;
  int b = id / cN, n = id % cN;
  float s = 0.f;
  unsigned short* row = catb + ((size_t)(b * cN + n)) * 192;
  for (int c = 0; c < 64; ++c) {
    float v = x[((size_t)(b * 64 + c)) * cN + n];
    s += v * v;
    row[c] = f2bf(v);
  }
  sq1[id] = s;
}

__global__ __launch_bounds__(256) void prep_w_kernel(
    const float* __restrict__ W1, const float* __restrict__ W2,
    const float* __restrict__ W3, const float* __restrict__ W4,
    unsigned short* __restrict__ wb) {
  int id = blockIdx.x * 256 + threadIdx.x;
  if (id < 8192)        wb[id] = f2bf(W1[id]);
  else if (id < 12288)  wb[id] = f2bf(W2[id - 8192]);
  else if (id < 20480)  wb[id] = f2bf(W3[id - 12288]);
  else if (id < 45056)  wb[id] = f2bf(W4[id - 20480]);
}

// ---------------------------------------------------------------------------
// Fused kNN: one wave per 16-point strip.  WMMA computes 16x16 inner-product
// tiles (K=64 channels in two 16x16x32 bf16 steps); lanes 0..15 each own one
// row and maintain a register-resident top-20 list of pd = 2*ip - |xi|^2 -
// |xj|^2.  No NxN distance matrix is materialized.
// ---------------------------------------------------------------------------
__global__ __launch_bounds__(32) void knn_topk_kernel(
    const unsigned short* __restrict__ pts, int ch_off,
    const float* __restrict__ sqn, int* __restrict__ idx_out) {
  const int lane  = threadIdx.x;
  const int b     = blockIdx.y;
  const int strip = blockIdx.x;                       // 0..255
  const int m  = lane & 15;
  const int hi = lane >> 4;
  const int row_pt = strip * 16 + m;

  const unsigned short* rowp =
      pts + ((size_t)(b * cN + row_pt)) * 192 + ch_off;
  // A fragments (rows = strip points, K = channels)
  v16bf a0 = ld_frag16(rowp +      16 * hi, rowp + 8 + 16 * hi);   // ch 0..31
  v16bf a1 = ld_frag16(rowp + 32 + 16 * hi, rowp + 40 + 16 * hi);   // ch 32..63
  float sqrow = sqn[b * cN + row_pt];

  __shared__ float tile[16][17];
  __shared__ float sqt[16];

  float best[cK];
  int   bidx[cK];
#pragma unroll
  for (int t = 0; t < cK; ++t) { best[t] = -3.4e38f; bidx[t] = 0; }
  float vmin = -3.4e38f;

  for (int ct = 0; ct < cN / 16; ++ct) {
    const int colbase = ct * 16;
    const unsigned short* colp =
        pts + ((size_t)(b * cN + colbase + m)) * 192 + ch_off + 16 * hi;
    v16bf b0 = ld_frag16(colp,      colp + 8);        // ch 0..31
    v16bf b1 = ld_frag16(colp + 32, colp + 40);       // ch 32..63
    v8f c = {};
    c = wmma_bf16(a0, b0, c);
    c = wmma_bf16(a1, b1, c);
#pragma unroll
    for (int r = 0; r < 8; ++r) tile[r + 8 * hi][m] = c[r];
    if (lane < 16) sqt[lane] = sqn[b * cN + colbase + lane];
    __syncthreads();
    if (lane < 16) {
      for (int j = 0; j < 16; ++j) {
        float pd = 2.0f * tile[lane][j] - sqrow - sqt[j];
        if (pd > vmin) {
          bool done = false;
#pragma unroll
          for (int t = 0; t < cK; ++t)
            if (!done && best[t] == vmin) {
              best[t] = pd; bidx[t] = colbase + j; done = true;
            }
          vmin = best[0];
#pragma unroll
          for (int t = 1; t < cK; ++t) vmin = fminf(vmin, best[t]);
        }
      }
    }
    __syncthreads();
  }
  if (lane < 16) {
    int* o = idx_out + ((size_t)(b * cN + row_pt)) * cK;
#pragma unroll
    for (int t = 0; t < cK; ++t) o[t] = bidx[t];
  }
}

// ---------------------------------------------------------------------------
// Fused graph-feature + GEMM:  y(64 x NK) = W(64x128) * F(128 x NK), where the
// F column for position (n,k) is [x[idx]-x[n] ; x[n]] built on the fly in
// registers (never materialized).  One wave computes a 64x16 output tile via
// 4 o-tiles x 4 K-chunks of bf16 WMMA.
// ---------------------------------------------------------------------------
__global__ __launch_bounds__(32) void conv_gf_kernel(
    const unsigned short* __restrict__ Wb,   // [64][128] bf16
    const unsigned short* __restrict__ pts,  // catb
    int ch_off,
    const int* __restrict__ idx,
    float* __restrict__ y) {                 // [b][64][NKp]
  const int lane = threadIdx.x;
  const int b    = blockIdx.y;
  const int tile = blockIdx.x;               // 0..NKp/16-1
  const int m  = lane & 15;
  const int hi = lane >> 4;
  const int col = tile * 16 + m;
  const int n  = col / cK;
  const int kk = col - n * cK;
  const int nb = idx[(b * cN + n) * cK + kk];

  const unsigned short* xc = pts + ((size_t)(b * cN + n )) * 192 + ch_off;
  const unsigned short* xn = pts + ((size_t)(b * cN + nb)) * 192 + ch_off;

  v16bf bf[4];
#pragma unroll
  for (int kc = 0; kc < 2; ++kc) {           // channels 0..63: neigh - center
    H16 s;
    const int base = 16 * hi + 32 * kc;
#pragma unroll
    for (int j = 0; j < 16; ++j)
      s.h[j] = f2bf(bf2f(xn[base + j]) - bf2f(xc[base + j]));
    bf[kc] = __builtin_bit_cast(v16bf, s);
  }
#pragma unroll
  for (int kc = 2; kc < 4; ++kc) {           // channels 64..127: center
    const unsigned short* p = xc + 16 * hi + 32 * (kc - 2);
    bf[kc] = ld_frag16(p, p + 8);
  }

  for (int ot = 0; ot < 4; ++ot) {
    const unsigned short* wrow = Wb + (ot * 16 + m) * 128;
    v8f c = {};
#pragma unroll
    for (int kc = 0; kc < 4; ++kc) {
      v16bf a = ld_frag16(wrow + 32 * kc + 16 * hi,
                          wrow + 32 * kc + 8 + 16 * hi);
      c = wmma_bf16(a, bf[kc], c);
    }
#pragma unroll
    for (int r = 0; r < 8; ++r) {
      int o = ot * 16 + r + 8 * hi;
      y[((size_t)(b * 64 + o)) * NKp + col] = c[r];
    }
  }
}

// ---------------------------------------------------------------------------
// Dense GEMM: y(nout x cols) = W(nout x nin) * in(nin x cols); input is bf16
// column-major-per-position (contiguous channels per column).
// ---------------------------------------------------------------------------
template <int KCS, int OTS>
__global__ __launch_bounds__(32) void conv_dense_kernel(
    const unsigned short* __restrict__ Wb,   // [OTS*16][KCS*32] bf16
    const unsigned short* __restrict__ in,   // [b][cols][in_stride] bf16
    int in_stride, int in_off, int cols,
    float* __restrict__ y) {                 // [b][OTS*16][cols]
  const int lane = threadIdx.x;
  const int b    = blockIdx.y;
  const int tile = blockIdx.x;
  const int m  = lane & 15;
  const int hi = lane >> 4;
  const int col = tile * 16 + m;

  const unsigned short* cp =
      in + ((size_t)b * cols + col) * in_stride + in_off;
  v16bf bf[KCS];
#pragma unroll
  for (int kc = 0; kc < KCS; ++kc) {
    const unsigned short* p = cp + 16 * hi + 32 * kc;
    bf[kc] = ld_frag16(p, p + 8);
  }
#pragma unroll
  for (int ot = 0; ot < OTS; ++ot) {
    const unsigned short* wrow = Wb + (ot * 16 + m) * (KCS * 32);
    v8f c = {};
#pragma unroll
    for (int kc = 0; kc < KCS; ++kc) {
      v16bf a = ld_frag16(wrow + 32 * kc + 16 * hi,
                          wrow + 32 * kc + 8 + 16 * hi);
      c = wmma_bf16(a, bf[kc], c);
    }
#pragma unroll
    for (int r = 0; r < 8; ++r) {
      int o = ot * 16 + r + 8 * hi;
      y[((size_t)(b * OTS * 16 + o)) * cols + col] = c[r];
    }
  }
}

// ---------------------------------------------------------------------------
// Per-channel sum / sum-of-squares over [b][nch][M] (one WG per channel).
// ---------------------------------------------------------------------------
__global__ __launch_bounds__(256) void stats_kernel(
    const float* __restrict__ y, float* __restrict__ stats, int nch, int M) {
  const int c = blockIdx.x, tid = threadIdx.x;
  float s = 0.f, s2 = 0.f;
  for (int b = 0; b < cB; ++b) {
    const float* p = y + ((size_t)(b * nch + c)) * M;
    for (int i = tid; i < M; i += 256) { float v = p[i]; s += v; s2 += v * v; }
  }
  __shared__ float ls[256], ls2[256];
  ls[tid] = s; ls2[tid] = s2;
  __syncthreads();
  for (int off = 128; off > 0; off >>= 1) {
    if (tid < off) { ls[tid] += ls[tid + off]; ls2[tid] += ls2[tid + off]; }
    __syncthreads();
  }
  if (tid == 0) { stats[2 * c] = ls[0]; stats[2 * c + 1] = ls2[0]; }
}

// BN + LeakyReLU + transpose -> bf16 [b][p][64]  (feeds next dense GEMM)
__global__ __launch_bounds__(256) void bn_tr_kernel(
    const float* __restrict__ y, const float* __restrict__ stats,
    const float* __restrict__ g, const float* __restrict__ bt,
    unsigned short* __restrict__ zb) {
  size_t id = (size_t)blockIdx.x * 256 + threadIdx.x;   // B*64*NKp
  if (id >= (size_t)cB * 64 * NKp) return;
  int    p = (int)(id % NKp);
  size_t r = id / NKp;
  int    o = (int)(r % 64);
  int    b = (int)(r / 64);
  float cnt  = (float)cB * (float)NKp;
  float mean = stats[2 * o] / cnt;
  float var  = stats[2 * o + 1] / cnt - mean * mean;
  float z = (y[id] - mean) * rsqrtf(var + EPSc) * g[o] + bt[o];
  z = z >= 0.f ? z : SLP * z;
  zb[((size_t)(b * NKp + p)) * 64 + o] = f2bf(z);
}

// BN + LeakyReLU + max over k -> catb[ch_off..ch_off+63] (bf16), optional |x|^2
__global__ __launch_bounds__(256) void bn_max_kernel(
    const float* __restrict__ y, const float* __restrict__ stats,
    const float* __restrict__ g, const float* __restrict__ bt,
    unsigned short* __restrict__ catb, int ch_off,
    float* __restrict__ sqout) {
  int id = blockIdx.x * 256 + threadIdx.x;              // (b,n)
  if (id >= cB * cN) return;
  int b = id / cN, n = id % cN;
  float sq = 0.f;
  const float cnt = (float)cB * (float)NKp;
  unsigned short* row = catb + ((size_t)(b * cN + n)) * 192 + ch_off;
  for (int o = 0; o < 64; ++o) {
    float mean = stats[2 * o] / cnt;
    float var  = stats[2 * o + 1] / cnt - mean * mean;
    float sc = rsqrtf(var + EPSc) * g[o];
    float sh = bt[o] - mean * sc;
    const float* p = y + ((size_t)(b * 64 + o)) * NKp + n * cK;
    float mx = -3.4e38f;
#pragma unroll 4
    for (int k = 0; k < cK; ++k) {
      float z = p[k] * sc + sh;
      z = z >= 0.f ? z : SLP * z;
      mx = fmaxf(mx, z);
    }
    row[o] = f2bf(mx);
    sq += mx * mx;
  }
  if (sqout) sqout[id] = sq;
}

// Final BN + LeakyReLU -> f32 output (B, 128, N)
__global__ __launch_bounds__(256) void bn_out_kernel(
    const float* __restrict__ y, const float* __restrict__ stats,
    const float* __restrict__ g, const float* __restrict__ bt,
    float* __restrict__ out) {
  size_t id = (size_t)blockIdx.x * 256 + threadIdx.x;   // B*128*N
  if (id >= (size_t)cB * 128 * cN) return;
  int o = (int)((id / cN) % 128);
  float cnt  = (float)cB * (float)cN;
  float mean = stats[2 * o] / cnt;
  float var  = stats[2 * o + 1] / cnt - mean * mean;
  float z = (y[id] - mean) * rsqrtf(var + EPSc) * g[o] + bt[o];
  out[id] = z >= 0.f ? z : SLP * z;
}

// ---------------------------------------------------------------------------
extern "C" void kernel_launch(void* const* d_in, const int* in_sizes, int n_in,
                              void* d_out, int out_size, void* d_ws,
                              size_t ws_size, hipStream_t stream) {
  (void)in_sizes; (void)n_in; (void)out_size; (void)ws_size;
  const float* x  = (const float*)d_in[0];
  const float* W1 = (const float*)d_in[1];
  const float* g1 = (const float*)d_in[2];
  const float* b1 = (const float*)d_in[3];
  const float* W2 = (const float*)d_in[4];
  const float* g2 = (const float*)d_in[5];
  const float* b2 = (const float*)d_in[6];
  const float* W3 = (const float*)d_in[7];
  const float* g3 = (const float*)d_in[8];
  const float* b3 = (const float*)d_in[9];
  const float* W4 = (const float*)d_in[10];
  const float* g4 = (const float*)d_in[11];
  const float* b4 = (const float*)d_in[12];
  float* out = (float*)d_out;

  // Workspace carve-up (256B aligned).
  char*  base = (char*)d_ws;
  size_t off  = 0;
  auto carve = [&](size_t bytes) {
    char* p = base + off;
    off = (off + bytes + 255) & ~(size_t)255;
    return p;
  };
  unsigned short* catb = (unsigned short*)carve((size_t)cB * cN * 192 * 2);
  float*          sq1  = (float*)carve((size_t)cB * cN * 4);
  float*          sq2  = (float*)carve((size_t)cB * cN * 4);
  int*            idx1 = (int*)carve((size_t)cB * cN * cK * 4);
  int*            idx2 = (int*)carve((size_t)cB * cN * cK * 4);
  unsigned short* wb   = (unsigned short*)carve((size_t)45056 * 2);
  float*          ybuf = (float*)carve((size_t)cB * 64 * NKp * 4);
  unsigned short* zb   = (unsigned short*)carve((size_t)cB * NKp * 64 * 2);
  float*          stats = (float*)carve((size_t)128 * 2 * 4);

  unsigned short* w1b = wb;
  unsigned short* w2b = wb + 8192;
  unsigned short* w3b = wb + 12288;
  unsigned short* w4b = wb + 20480;

  const dim3 wmma_blk(32);
  const dim3 knn_grid(cN / 16, cB);
  const dim3 conv_grid(NKp / 16, cB);
  const dim3 conv4_grid(cN / 16, cB);

  // Stage 0: precision staging
  prep_x_kernel<<<(cB * cN) / 256, 256, 0, stream>>>(x, catb, sq1);
  prep_w_kernel<<<176, 256, 0, stream>>>(W1, W2, W3, W4, wb);

  // Stage 1: knn on x1, edge-conv1, BN+lrelu -> zb
  knn_topk_kernel<<<knn_grid, wmma_blk, 0, stream>>>(catb, 0, sq1, idx1);
  conv_gf_kernel<<<conv_grid, wmma_blk, 0, stream>>>(w1b, catb, 0, idx1, ybuf);
  stats_kernel<<<64, 256, 0, stream>>>(ybuf, stats, 64, NKp);
  bn_tr_kernel<<<(int)(((size_t)cB * 64 * NKp) / 256), 256, 0, stream>>>(
      ybuf, stats, g1, b1, zb);

  // Stage 2: conv2 (64x64), BN+lrelu+max_k -> x2 (catb[64..127]) + |x2|^2
  conv_dense_kernel<2, 4><<<conv_grid, wmma_blk, 0, stream>>>(
      w2b, zb, 64, 0, NKp, ybuf);
  stats_kernel<<<64, 256, 0, stream>>>(ybuf, stats, 64, NKp);
  bn_max_kernel<<<(cB * cN) / 256, 256, 0, stream>>>(
      ybuf, stats, g2, b2, catb, 64, sq2);

  // Stage 3: knn on x2, edge-conv3, BN+lrelu+max_k -> x3 (catb[128..191])
  knn_topk_kernel<<<knn_grid, wmma_blk, 0, stream>>>(catb, 64, sq2, idx2);
  conv_gf_kernel<<<conv_grid, wmma_blk, 0, stream>>>(w3b, catb, 64, idx2, ybuf);
  stats_kernel<<<64, 256, 0, stream>>>(ybuf, stats, 64, NKp);
  bn_max_kernel<<<(cB * cN) / 256, 256, 0, stream>>>(
      ybuf, stats, g3, b3, catb, 128, nullptr);

  // Stage 4: conv4 (128x192) over cat=[x1;x2;x3], BN+lrelu -> out
  conv_dense_kernel<6, 8><<<conv4_grid, wmma_blk, 0, stream>>>(
      w4b, catb, 192, 0, cN, ybuf);
  stats_kernel<<<128, 256, 0, stream>>>(ybuf, stats, 128, cN);
  bn_out_kernel<<<(int)(((size_t)cB * 128 * cN) / 256), 256, 0, stream>>>(
      ybuf, stats, g4, b4, out);
}
